// DGCNN_58153857188560
// MI455X (gfx1250) — compile-verified, matrix operand order checked by reading, not run on hardware
//
#include <hip/hip_runtime.h>

// ---------------------------------------------------------------------------
// DGCNN forward for MI455X (gfx1250, wave32, WMMA bf16 16x16x32).
// B=4, C=3, N=8192, K=5.  Pipeline:
//   (A) cvt:    fp32 weights -> bf16 (w2,w3,w4,w5)
//   (B) knn:    top-5 neighbor indices per point (x chunks in LDS, reg top-5)
//   (C) fused:  conv1(VALU) + conv2/3/4 (WMMA bf16) per 16-point tile in LDS,
//               per-point max-over-k done in accumulator registers
//               (column mapping col = k*16 + p), writes cat (B,N,512) bf16
//   (D) conv5:  512x512 GEMM over 32768 columns, 32 cols/block; cat tile is
//               DMA'd to LDS by the Tensor Data Mover (padded layout via
//               D#.pad_enable), WMMA bf16, relu, fp32 out
// ---------------------------------------------------------------------------

typedef unsigned short u16;
typedef unsigned int   u32;
typedef __attribute__((ext_vector_type(16))) __bf16 v16bf;
typedef __attribute__((ext_vector_type(8)))  float  v8f;
typedef __attribute__((ext_vector_type(4)))  u32    uv4;
typedef __attribute__((ext_vector_type(8)))  int    iv8;
typedef __attribute__((ext_vector_type(4)))  int    iv4;

constexpr int SN  = 8192;  // points per batch
constexpr int SB  = 4;     // batches
constexpr int KNN = 5;
constexpr int C5COLS = 32; // conv5 columns per block

// workspace layout (bytes)
constexpr size_t OFF_IDX = 0;                           // 4*8192*5*4  = 655360
constexpr size_t OFF_W2B = 655360;                      // 64*64*2     = 8192
constexpr size_t OFF_W3B = OFF_W2B + 8192;              // 128*64*2    = 16384
constexpr size_t OFF_W4B = OFF_W3B + 16384;             // 256*128*2   = 65536
constexpr size_t OFF_W5B = OFF_W4B + 65536;             // 512*512*2   = 524288
constexpr size_t OFF_CAT = OFF_W5B + 524288;            // 4*8192*512*2 = 32MB

union Frag {
    v16bf v;
    u16   h[16];
    uv4   q[2];
};

__device__ __forceinline__ u16 f2bf(float f) {
    u32 u = __float_as_uint(f);
    u += 0x7FFFu + ((u >> 16) & 1u);       // round-to-nearest-even
    return (u16)(u >> 16);
}

__device__ __forceinline__ v8f zero8() {
    v8f z;
    #pragma unroll
    for (int i = 0; i < 8; ++i) z[i] = 0.0f;
    return z;
}

// A/B fragment load: two contiguous 16B chunks at K = kbase+g*8 and +16.
// Caller passes p already offset by row*stride + kbase + g*8.
__device__ __forceinline__ void load_frag(Frag* f, const u16* p) {
    f->q[0] = *(const uv4*)(p);
    f->q[1] = *(const uv4*)(p + 16);
}

// ---------------------------------------------------------------------------
// (A) weight conversion fp32 -> bf16
// ---------------------------------------------------------------------------
__global__ __launch_bounds__(256) void cvt_kernel(
    const float* __restrict__ w2, const float* __restrict__ w3,
    const float* __restrict__ w4, const float* __restrict__ w5,
    u16* __restrict__ w2b, u16* __restrict__ w3b,
    u16* __restrict__ w4b, u16* __restrict__ w5b)
{
    int t = blockIdx.x * 256 + threadIdx.x;
    if (t < 64 * 64)    w2b[t] = f2bf(w2[t]);
    if (t < 128 * 64)   w3b[t] = f2bf(w3[t]);
    if (t < 256 * 128)  w4b[t] = f2bf(w4[t]);
    if (t < 512 * 512)  w5b[t] = f2bf(w5[t]);
}

// ---------------------------------------------------------------------------
// (B) kNN: each thread owns one query point, scans all candidates in 32KB
// LDS chunks (uniform j -> LDS broadcast), keeps sorted top-5 in registers.
// Strict '<' keeps the lowest index on ties, matching jax.lax.top_k.
// ---------------------------------------------------------------------------
__global__ __launch_bounds__(256) void knn_kernel(
    const float* __restrict__ x, int* __restrict__ idxp)
{
    __shared__ float4 pts[2048];
    const int tid = threadIdx.x;
    const int b   = blockIdx.y;
    const int q   = blockIdx.x * 256 + tid;
    const float* xb = x + (size_t)b * 3 * SN;

    const float qx = xb[q], qy = xb[SN + q], qz = xb[2 * SN + q];

    float d0 = 3e38f, d1 = 3e38f, d2 = 3e38f, d3 = 3e38f, d4 = 3e38f;
    int   i0 = 0, i1 = 0, i2 = 0, i3 = 0, i4 = 0;

    for (int ch = 0; ch < SN / 2048; ++ch) {
        __syncthreads();
        for (int i = tid; i < 2048; i += 256) {
            int j = ch * 2048 + i;
            pts[i] = make_float4(xb[j], xb[SN + j], xb[2 * SN + j], 0.0f);
        }
        __syncthreads();
        const int jbase = ch * 2048;
        for (int i = 0; i < 2048; ++i) {
            float4 p = pts[i];                       // uniform addr -> broadcast
            float dx = p.x - qx, dy = p.y - qy, dz = p.z - qz;
            float d  = dx * dx + dy * dy + dz * dz;  // minimize ||xi-xj||^2
            int   j  = jbase + i;
            if (d < d4) {
                if (d < d0)      { d4=d3;i4=i3; d3=d2;i3=i2; d2=d1;i2=i1; d1=d0;i1=i0; d0=d;i0=j; }
                else if (d < d1) { d4=d3;i4=i3; d3=d2;i3=i2; d2=d1;i2=i1; d1=d;i1=j; }
                else if (d < d2) { d4=d3;i4=i3; d3=d2;i3=i2; d2=d;i2=j; }
                else if (d < d3) { d4=d3;i4=i3; d3=d;i3=j; }
                else             { d4=d;i4=j; }
            }
        }
    }
    int* op = idxp + (size_t)(b * SN + q) * KNN;
    op[0] = i0; op[1] = i1; op[2] = i2; op[3] = i3; op[4] = i4;
}

// ---------------------------------------------------------------------------
// GEMM stage: hout[COUT x 80cols] = relu(W[COUT x CIN] * hin[CIN x 80cols]),
// bf16 WMMA, fp32 acc. Column mapping col = k*16 + p, so the 5 column-tile
// accumulators hold the same 16 points at k=0..4 -> max-over-k is a register
// elementwise max. SH: store relu'd h to LDS (bf16). SC: store per-point max
// (relu'd) to cat (global bf16, [point][channel]).
// ---------------------------------------------------------------------------
template<int CIN, int COUT, bool SH, bool SC>
__device__ __forceinline__ void gemm_stage(
    const u16* __restrict__ W, const u16* hin, int sin,
    u16* hout, int sout, u16* __restrict__ catp, int lane)
{
    const int g = lane >> 4, r = lane & 15;
    #pragma unroll
    for (int mt = 0; mt < COUT / 16; ++mt) {
        v8f acc[5];
        #pragma unroll
        for (int t = 0; t < 5; ++t) acc[t] = zero8();
        #pragma unroll
        for (int kt = 0; kt < CIN / 32; ++kt) {
            Frag a;
            load_frag(&a, W + (size_t)(mt * 16 + r) * CIN + kt * 32 + g * 8);
            #pragma unroll
            for (int t = 0; t < 5; ++t) {
                Frag bf;
                load_frag(&bf, hin + (t * 16 + r) * sin + kt * 32 + g * 8);
                acc[t] = __builtin_amdgcn_wmma_f32_16x16x32_bf16(
                    false, a.v, false, bf.v, (short)0, acc[t], false, false);
            }
        }
        if constexpr (SH) {
            #pragma unroll
            for (int t = 0; t < 5; ++t) {
                uv4 pk;
                #pragma unroll
                for (int i = 0; i < 4; ++i)
                    pk[i] = (u32)f2bf(fmaxf(acc[t][2 * i], 0.0f)) |
                            ((u32)f2bf(fmaxf(acc[t][2 * i + 1], 0.0f)) << 16);
                *(uv4*)(hout + (t * 16 + r) * sout + mt * 16 + g * 8) = pk;
            }
        }
        if constexpr (SC) {
            v8f xm = acc[0];
            #pragma unroll
            for (int t = 1; t < 5; ++t)
                #pragma unroll
                for (int i = 0; i < 8; ++i) xm[i] = fmaxf(xm[i], acc[t][i]);
            uv4 pk;
            #pragma unroll
            for (int i = 0; i < 4; ++i)
                pk[i] = (u32)f2bf(fmaxf(xm[2 * i], 0.0f)) |
                        ((u32)f2bf(fmaxf(xm[2 * i + 1], 0.0f)) << 16);
            *(uv4*)(catp + (size_t)r * 512 + mt * 16 + g * 8) = pk;
        }
    }
}

// ---------------------------------------------------------------------------
// (C) fused conv1..conv4, one wave per block, 16 points (80 columns).
// ---------------------------------------------------------------------------
__global__ __launch_bounds__(32) void edgeconv_kernel(
    const float* __restrict__ x, const int* __restrict__ idxp,
    const float* __restrict__ w1,
    const u16* __restrict__ w2b, const u16* __restrict__ w3b,
    const u16* __restrict__ w4b, u16* __restrict__ catb)
{
    constexpr int S1 = 72;    // 64ch + 8 pad (shorts) -> bank-conflict free
    constexpr int S3 = 136;   // 128ch + 8 pad
    __shared__ float w1s[64 * 6];
    __shared__ u16 h1[80 * S1];
    __shared__ u16 h2[80 * S1];
    __shared__ u16 h3[80 * S3];

    const int lane = threadIdx.x;
    const int b    = blockIdx.x >> 9;          // 512 blocks per batch
    const int n0   = (blockIdx.x & 511) << 4;  // 16 points per block
    const float* xb = x + (size_t)b * 3 * SN;
    u16* catp = catb + ((size_t)(b * SN + n0)) * 512;

    for (int i = lane; i < 64 * 6; i += 32) w1s[i] = w1[i];
    __syncthreads();

    // stage 1: feat(6) -> h1(64), VALU (K too small for WMMA)
    for (int col = lane; col < 80; col += 32) {
        int p = col & 15, k = col >> 4;
        int n = n0 + p;
        int j = idxp[(size_t)(b * SN + n) * KNN + k];
        float f0 = xb[j], f1 = xb[SN + j], f2 = xb[2 * SN + j];
        float f3 = xb[n], f4 = xb[SN + n], f5 = xb[2 * SN + n];
        u16* dst = h1 + col * S1;
        #pragma unroll
        for (int oc = 0; oc < 64; oc += 8) {
            uv4 pk;
            #pragma unroll
            for (int i2 = 0; i2 < 4; ++i2) {
                const float* wr = w1s + (oc + 2 * i2) * 6;
                float a0 = wr[0]*f0 + wr[1]*f1 + wr[2]*f2 + wr[3]*f3 + wr[4]*f4 + wr[5]*f5;
                float a1 = wr[6]*f0 + wr[7]*f1 + wr[8]*f2 + wr[9]*f3 + wr[10]*f4 + wr[11]*f5;
                pk[i2] = (u32)f2bf(fmaxf(a0, 0.0f)) | ((u32)f2bf(fmaxf(a1, 0.0f)) << 16);
            }
            *(uv4*)(dst + oc) = pk;
        }
    }
    __syncthreads();

    // x1 = max_k h1 -> cat[0:64]  (post-relu bf16 >= 0 -> bit-pattern max OK)
    for (int t = lane; t < 16 * 64; t += 32) {
        int p = t >> 6, c = t & 63;
        u16 m = 0;
        #pragma unroll
        for (int k = 0; k < 5; ++k) {
            u16 v = h1[(k * 16 + p) * S1 + c];
            m = v > m ? v : m;
        }
        catp[(size_t)p * 512 + c] = m;
    }
    __syncthreads();

    gemm_stage< 64,  64, true,  true >(w2b, h1, S1, h2, S1, catp + 64,  lane);
    __syncthreads();
    gemm_stage< 64, 128, true,  true >(w3b, h2, S1, h3, S3, catp + 128, lane);
    __syncthreads();
    gemm_stage<128, 256, false, true >(w4b, h3, S3, nullptr, 0, catp + 256, lane);
}

// ---------------------------------------------------------------------------
// (D) conv5: out(512 x 32768cols) = relu(W5(512x512) * cat), fp32 out.
// 8 waves, 32 columns per block. Cat tile (32KB contiguous) is DMA'd into a
// padded LDS layout (stride 520 shorts) by the Tensor Data Mover:
// data_size=4B, tile 256x32 dwords, pad_interval=7 (every 256 dwords),
// pad_amount=3 (4 dwords = 16B). Each wave: 64 rows x 32 cols; per k-step
// 4 A-frag loads feed 8 WMMAs.
// ---------------------------------------------------------------------------
__global__ __launch_bounds__(256) void conv5_kernel(
    const u16* __restrict__ w5b, const u16* __restrict__ catb,
    float* __restrict__ out)
{
    constexpr int ST = 520;                  // 512 + 8 pad (shorts)
    __shared__ u16 tile[C5COLS * ST];        // 33,280 B
    const int tid  = threadIdx.x;
    const int lane = tid & 31, wave = tid >> 5;
    const int colbase = blockIdx.x * C5COLS; // over B*N columns

#if defined(__HIP_DEVICE_COMPILE__) && __has_builtin(__builtin_amdgcn_tensor_load_to_lds)
    if (wave == 0) {                         // one TDM DMA per block (EXEC ignored)
        unsigned long long ga =
            (unsigned long long)(const void*)catb + (unsigned long long)colbase * 1024ull;
        u32 lds = (u32)(uintptr_t)(&tile[0]);
        uv4 g0;
        g0[0] = 1u;                                  // count=1 (valid user D#)
        g0[1] = lds;                                 // lds_addr (bytes)
        g0[2] = (u32)(ga & 0xffffffffu);             // global_addr[31:0]
        g0[3] = (u32)(ga >> 32) | (2u << 30);        // global_addr[56:32] | type=2
        iv8 g1;
        g1[0] = (int)((2u << 16)      // data_size = 4B
                    | (1u << 20)      // pad_enable
                    | (7u << 22)      // pad_interval: 256 dwords
                    | (3u << 25));    // pad_amount:   4 dwords (16B)
        g1[1] = (int)(256u << 16);    // tensor_dim0 = 256 dwords/col
        g1[2] = (int)((u32)C5COLS << 16); // tensor_dim1 (low 16b at [95:80])
        g1[3] = (int)(256u << 16);    // tile_dim0 = 256
        g1[4] = (int)C5COLS;          // tile_dim1 = 32
        g1[5] = 256;                  // tensor_dim0_stride (contiguous)
        g1[6] = 0;
        g1[7] = 0;
        iv4 gz = {0, 0, 0, 0};        // 2D tensor: groups 2/3 unused
        iv8 gz8 = {0, 0, 0, 0, 0, 0, 0, 0};
        __builtin_amdgcn_tensor_load_to_lds(g0, g1, gz, gz, gz8, 0);
        __builtin_amdgcn_s_wait_tensorcnt(0);
    }
#else
    {
        const u32* src = (const u32*)catb;   // 256 dwords per column
        for (int i = tid; i < C5COLS * 256; i += 256) {
            int c = i >> 8, k2 = i & 255;
            *(u32*)(tile + c * ST + k2 * 2) = src[(size_t)(colbase + c) * 256 + k2];
        }
    }
#endif
    __syncthreads();

    const int g = lane >> 4, r = lane & 15;
    const int mbase = wave << 6;
    v8f acc[4][2];
    #pragma unroll
    for (int mt = 0; mt < 4; ++mt) {
        acc[mt][0] = zero8();
        acc[mt][1] = zero8();
    }

    for (int kt = 0; kt < 16; ++kt) {
        Frag bf0, bf1;
        load_frag(&bf0, tile + r * ST + kt * 32 + g * 8);
        load_frag(&bf1, tile + (16 + r) * ST + kt * 32 + g * 8);
        #pragma unroll
        for (int mt = 0; mt < 4; ++mt) {
            Frag a;
            load_frag(&a, w5b + (size_t)(mbase + mt * 16 + r) * 512 + kt * 32 + g * 8);
            acc[mt][0] = __builtin_amdgcn_wmma_f32_16x16x32_bf16(
                false, a.v, false, bf0.v, (short)0, acc[mt][0], false, false);
            acc[mt][1] = __builtin_amdgcn_wmma_f32_16x16x32_bf16(
                false, a.v, false, bf1.v, (short)0, acc[mt][1], false, false);
        }
    }

    const int b = colbase >> 13;
    #pragma unroll
    for (int t = 0; t < 2; ++t) {
        const int n = (colbase & (SN - 1)) + t * 16 + r;
        float* ob = out + (size_t)b * 512 * SN + n;
        #pragma unroll
        for (int mt = 0; mt < 4; ++mt) {
            #pragma unroll
            for (int i = 0; i < 8; ++i) {
                int o = mbase + mt * 16 + g * 8 + i;
                ob[(size_t)o * SN] = fmaxf(acc[mt][t][i], 0.0f);
            }
        }
    }
}

// ---------------------------------------------------------------------------
extern "C" void kernel_launch(void* const* d_in, const int* in_sizes, int n_in,
                              void* d_out, int out_size, void* d_ws, size_t ws_size,
                              hipStream_t stream) {
    (void)in_sizes; (void)n_in; (void)out_size; (void)ws_size;
    const float* x  = (const float*)d_in[0];
    const float* w1 = (const float*)d_in[1];
    const float* w2 = (const float*)d_in[2];
    const float* w3 = (const float*)d_in[3];
    const float* w4 = (const float*)d_in[4];
    const float* w5 = (const float*)d_in[5];

    char* ws   = (char*)d_ws;
    int*  idxp = (int*)(ws + OFF_IDX);
    u16*  w2b  = (u16*)(ws + OFF_W2B);
    u16*  w3b  = (u16*)(ws + OFF_W3B);
    u16*  w4b  = (u16*)(ws + OFF_W4B);
    u16*  w5b  = (u16*)(ws + OFF_W5B);
    u16*  catb = (u16*)(ws + OFF_CAT);

    cvt_kernel<<<(512 * 512 + 255) / 256, 256, 0, stream>>>(
        w2, w3, w4, w5, w2b, w3b, w4b, w5b);
    knn_kernel<<<dim3(SN / 256, SB), 256, 0, stream>>>(x, idxp);
    edgeconv_kernel<<<SB * (SN / 16), 32, 0, stream>>>(
        x, idxp, w1, w2b, w3b, w4b, catb);
    conv5_kernel<<<(SB * SN) / C5COLS, 256, 0, stream>>>(
        w5b, catb, (float*)d_out);
}
